// DoRALlamaAttention_26645977104864
// MI455X (gfx1250) — matrix operand order repaired and checked by reading, not compiled, not run on hardware
//
#include <hip/hip_runtime.h>

// ---------------------------------------------------------------------------
// DoRA Llama attention for MI455X (gfx1250, wave32, WMMA).
// All matmuls run on v_wmma_f32_16x16x32_f16 (f16 in, f32 accum).
// Projection GEMMs use double-buffered async global->LDS staging (ASYNCcnt).
// ---------------------------------------------------------------------------

typedef _Float16 half_t;
typedef __attribute__((ext_vector_type(16))) _Float16 v16h;
typedef __attribute__((ext_vector_type(8)))  float    v8f;

#define HIDDEN 4096
#define SEQ    2048
#define BATCH  2
#define NHEADS 32
#define NKVH   8
#define HDIM   128
#define KV_OUT (NKVH * HDIM)   // 1024
#define DORA_SCALING 2.0f      // alpha/rank = 16/8
#define DORA_EPS 1e-8f

#if defined(__HIP_DEVICE_COMPILE__) && __has_builtin(__builtin_amdgcn_global_load_async_to_lds_b128)
#define HAVE_ASYNC_LDS 1
#endif

// Pointer types matching the builtin's signature:
//   (int4 addrspace(1)* src, int4 addrspace(3)* dst, Ii offset, Ii cpol)
typedef int vi4 __attribute__((vector_size(16)));
typedef __attribute__((address_space(1))) vi4* gvi4p;
typedef __attribute__((address_space(3))) vi4* lvi4p;

// 16-byte global -> LDS copy. Async (ASYNCcnt-tracked, LDS-direct) when the
// gfx1250 builtin is available, else load+store through VGPRs.
__device__ __forceinline__ void cp16_to_lds(const half_t* g, half_t* l) {
#if HAVE_ASYNC_LDS
  __builtin_amdgcn_global_load_async_to_lds_b128(
      (gvi4p)const_cast<half_t*>(g), (lvi4p)l, 0, 0);
#else
  *reinterpret_cast<uint4*>(l) = *reinterpret_cast<const uint4*>(g);
#endif
}

template <int N>
__device__ __forceinline__ void wait_asynccnt() {
#if HAVE_ASYNC_LDS
#if __has_builtin(__builtin_amdgcn_s_wait_asynccnt)
  __builtin_amdgcn_s_wait_asynccnt(N);
#else
  asm volatile("s_wait_asynccnt %0" :: "i"(N) : "memory");
#endif
#endif
}

__device__ __forceinline__ v8f wmma_f16(v16h a, v16h b, v8f c) {
  // D = A(16x32) * B(32x16) + C, f32 accumulate
  return __builtin_amdgcn_wmma_f32_16x16x32_f16(
      /*neg_a=*/false, a, /*neg_b=*/false, b,
      /*c_mod=*/(short)0, c, /*reuse_a=*/false, /*reuse_b=*/false);
}

// A-matrix fragment (16x32 f16). This code's contiguous k order: lane l holds row m=l&15;
// element e -> k = e + 16*(l>>4). 16 contiguous halves.
__device__ __forceinline__ v16h load_a_frag(const half_t* rowp, int hi) {
  union { uint4 u[2]; v16h v; } t;
  t.u[0] = *reinterpret_cast<const uint4*>(rowp + 16 * hi);
  t.u[1] = *reinterpret_cast<const uint4*>(rowp + 16 * hi + 8);
  return t.v;
}

// B-matrix fragment (32x16 f16). Lane l holds col n=l&15; element e ->
// k = e + 16*(l>>4): 16 contiguous halves (caller pre-applies the +16*hi).
__device__ __forceinline__ v16h load_b_frag(const half_t* p) {
  union { uint4 u[2]; v16h v; } t;
  t.u[0] = *reinterpret_cast<const uint4*>(p);
  t.u[1] = *reinterpret_cast<const uint4*>(p + 8);
  return t.v;
}

// ---------------------------------------------------------------------------
// f32 -> f16 convert (activations)
// ---------------------------------------------------------------------------
__global__ void f32_to_f16_kernel(const float* __restrict__ x,
                                  half_t* __restrict__ y, int n) {
  int i = blockIdx.x * blockDim.x + threadIdx.x;
  if (i < n) y[i] = (half_t)x[i];
}

// ---------------------------------------------------------------------------
// DoRA weight prep: Wd[row,:] = (W[row,:] + 2*(B@A)[row,:]) * m/(||row||+eps)
// One block (256 thr) per output row; IN is always 4096.
// ---------------------------------------------------------------------------
__global__ __launch_bounds__(256)
void dora_wd_kernel(const float* __restrict__ W, const float* __restrict__ A,
                    const float* __restrict__ Bw, const float* __restrict__ mvec,
                    half_t* __restrict__ Wd) {
  const int row = blockIdx.x;
  __shared__ __align__(16) float we_s[HIDDEN];
  __shared__ float red[8];

  float br[8];
#pragma unroll
  for (int r = 0; r < 8; r++) br[r] = Bw[row * 8 + r];

  float ss = 0.f;
  for (int i = threadIdx.x; i < HIDDEN; i += 256) {
    float lora = 0.f;
#pragma unroll
    for (int r = 0; r < 8; r++) lora += br[r] * A[r * HIDDEN + i];
    float we = W[(size_t)row * HIDDEN + i] + DORA_SCALING * lora;
    we_s[i] = we;
    ss += we * we;
  }
#pragma unroll
  for (int off = 16; off; off >>= 1) ss += __shfl_xor(ss, off, 32);
  if ((threadIdx.x & 31) == 0) red[threadIdx.x >> 5] = ss;
  __syncthreads();
  if (threadIdx.x == 0) {
    float tot = 0.f;
#pragma unroll
    for (int i = 0; i < 8; i++) tot += red[i];
    red[0] = mvec[row] / (sqrtf(tot) + DORA_EPS);
  }
  __syncthreads();
  const float scale = red[0];
  for (int i = threadIdx.x; i < HIDDEN; i += 256)
    Wd[(size_t)row * HIDDEN + i] = (half_t)(we_s[i] * scale);
}

// ---------------------------------------------------------------------------
// GEMM: Y[M,N] = X[M,K] @ Wd[N,K]^T. 128x128 tile per block, 8 waves,
// each wave a 32x64 sub-tile (2x4 WMMA accumulators). Double-buffered
// K-step-32 staging through LDS via async global->LDS copies.
// ---------------------------------------------------------------------------
template <bool OUT_F32>
__global__ __launch_bounds__(256)
void gemm_xwT_kernel(const half_t* __restrict__ X, const half_t* __restrict__ Wd,
                     void* __restrict__ Yv, int M, int N, int K) {
  __shared__ __align__(16) half_t xT[2][128 * 40];  // 128 rows x 32 k (+8 pad)
  __shared__ __align__(16) half_t wT[2][128 * 40];

  const int tid  = threadIdx.x;
  const int lane = tid & 31, ln = lane & 15, hi = lane >> 4;
  const int wid  = tid >> 5, wm = wid >> 1, wn = wid & 1;
  const int mBase = blockIdx.y * 128, nBase = blockIdx.x * 128;

  const v8f vz = {};
  v8f acc[2][4];
#pragma unroll
  for (int i = 0; i < 2; i++)
#pragma unroll
    for (int j = 0; j < 4; j++) acc[i][j] = vz;

  const int lr = tid >> 1;            // 0..127 tile row
  const int lk = (tid & 1) * 16;      // 0 or 16 halves
  const half_t* xg = X  + (size_t)(mBase + lr) * K + lk;
  const half_t* wg = Wd + (size_t)(nBase + lr) * K + lk;

  auto stage = [&](int buf, int k0) {
    cp16_to_lds(xg + k0,     &xT[buf][lr * 40 + lk]);
    cp16_to_lds(xg + k0 + 8, &xT[buf][lr * 40 + lk + 8]);
    cp16_to_lds(wg + k0,     &wT[buf][lr * 40 + lk]);
    cp16_to_lds(wg + k0 + 8, &wT[buf][lr * 40 + lk + 8]);
  };

  const int nIter = K / 32;
  stage(0, 0);                        // prologue fill of buffer 0
  for (int it = 0; it < nIter; ++it) {
    const int cur = it & 1;
    __syncthreads();                  // everyone done reading buffer cur^1
    if (it + 1 < nIter) {
      stage(cur ^ 1, (it + 1) * 32);  // issue next tile (4 async b128/thread)
      wait_asynccnt<4>();             // drain the ops filling buffer 'cur'
    } else {
      wait_asynccnt<0>();
    }
    __syncthreads();                  // buffer 'cur' visible to all waves

    v16h a0 = load_a_frag(&xT[cur][(wm * 32 +      ln) * 40], hi);
    v16h a1 = load_a_frag(&xT[cur][(wm * 32 + 16 + ln) * 40], hi);
    v16h b0 = load_b_frag(&wT[cur][(wn * 64 +      ln) * 40 + 16 * hi]);
    v16h b1 = load_b_frag(&wT[cur][(wn * 64 + 16 + ln) * 40 + 16 * hi]);
    v16h b2 = load_b_frag(&wT[cur][(wn * 64 + 32 + ln) * 40 + 16 * hi]);
    v16h b3 = load_b_frag(&wT[cur][(wn * 64 + 48 + ln) * 40 + 16 * hi]);
    acc[0][0] = wmma_f16(a0, b0, acc[0][0]);
    acc[0][1] = wmma_f16(a0, b1, acc[0][1]);
    acc[0][2] = wmma_f16(a0, b2, acc[0][2]);
    acc[0][3] = wmma_f16(a0, b3, acc[0][3]);
    acc[1][0] = wmma_f16(a1, b0, acc[1][0]);
    acc[1][1] = wmma_f16(a1, b1, acc[1][1]);
    acc[1][2] = wmma_f16(a1, b2, acc[1][2]);
    acc[1][3] = wmma_f16(a1, b3, acc[1][3]);
  }

  // C/D layout: lane l -> n = l&15; element e -> m = e + 8*(l>>4).
#pragma unroll
  for (int i = 0; i < 2; i++)
#pragma unroll
    for (int j = 0; j < 4; j++)
#pragma unroll
      for (int e = 0; e < 8; e++) {
        int m = mBase + wm * 32 + i * 16 + e + 8 * hi;
        int n = nBase + wn * 64 + j * 16 + ln;
        if (OUT_F32)
          ((float*)Yv)[(size_t)m * N + n] = acc[i][j][e];
        else
          ((half_t*)Yv)[(size_t)m * N + n] = (half_t)acc[i][j][e];
      }
}

// ---------------------------------------------------------------------------
// Fused causal flash attention with GQA (32 Q heads over 8 KV heads).
// Block: (qTile, head, batch); 8 waves x 16 query rows = 128 rows/block.
// 32-key tiles; tiles above the diagonal skipped entirely; the additive mask
// is read only for diagonal-crossing tiles (it is exactly 0 below diagonal).
// ---------------------------------------------------------------------------
__global__ __launch_bounds__(256)
void flash_attn_kernel(const half_t* __restrict__ Q, const half_t* __restrict__ Kg,
                       const half_t* __restrict__ Vg, const float* __restrict__ mask,
                       half_t* __restrict__ O) {
  const int qTile = blockIdx.x, head = blockIdx.y, bz = blockIdx.z;
  const int kvh = head >> 2;  // 32 heads / 8 kv heads
  const int tid = threadIdx.x, w = tid >> 5, lane = tid & 31;
  const int ln = lane & 15, hi = lane >> 4;
  const int qRow0 = qTile * 128 + w * 16;

  __shared__ __align__(16) half_t Kl[32 * 136];       // [key][dim], pad 8
  __shared__ __align__(16) half_t Vt[128 * 40];       // [dim][key], pad 8
  __shared__ __align__(16) half_t Pl[8][16 * 40];     // per-wave P tile

  // Q fragments: A(16x32) over head-dim chunks of 32 (4 chunks for d=128)
  v16h qf[4];
  {
    const half_t* qb = Q + ((size_t)(bz * SEQ + qRow0 + ln)) * HIDDEN + head * HDIM;
#pragma unroll
    for (int kc = 0; kc < 4; kc++) qf[kc] = load_a_frag(qb + kc * 32, hi);
  }

  float run_max[8], run_sum[8];
  const v8f vz = {};
  v8f outa[8];
#pragma unroll
  for (int e = 0; e < 8; e++) { run_max[e] = -3e38f; run_sum[e] = 0.f; }
#pragma unroll
  for (int j = 0; j < 8; j++) outa[j] = vz;

  const float scale = 0.08838834764831845f;  // 1/sqrt(128)
  const int nkt = (qTile + 1) * 4;           // causal: keys <= qTile*128+127

  for (int kt = 0; kt < nkt; kt++) {
    const int key0 = kt * 32;
    __syncthreads();
    // Stage K (row-major) and V (transposed) tiles: 32 keys x 128 dims.
    {
      const int r = tid >> 3;             // key row 0..31
      const int doff = (tid & 7) * 16;    // dim offset
      const size_t kvoff = ((size_t)(bz * SEQ + key0 + r)) * KV_OUT + kvh * HDIM + doff;
      const half_t* ksrc = Kg + kvoff;
      *reinterpret_cast<uint4*>(&Kl[r * 136 + doff]) =
          *reinterpret_cast<const uint4*>(ksrc);
      *reinterpret_cast<uint4*>(&Kl[r * 136 + doff + 8]) =
          *reinterpret_cast<const uint4*>(ksrc + 8);
      const half_t* vsrc = Vg + kvoff;
      union { uint4 u[2]; half_t h[16]; } vv;
      vv.u[0] = *reinterpret_cast<const uint4*>(vsrc);
      vv.u[1] = *reinterpret_cast<const uint4*>(vsrc + 8);
#pragma unroll
      for (int i = 0; i < 16; i++) Vt[(doff + i) * 40 + r] = vv.h[i];
      if (kt + 1 < nkt) {  // prefetch next tile (global_prefetch_b8)
        __builtin_prefetch(Kg + kvoff + 32 * KV_OUT, 0, 0);
        __builtin_prefetch(Vg + kvoff + 32 * KV_OUT, 0, 0);
      }
    }
    __syncthreads();

    // Logits S = Q K^T for 16 rows x 32 keys (2 n-blocks).
    v8f sf[2];
    sf[0] = vz; sf[1] = vz;
#pragma unroll
    for (int nb = 0; nb < 2; nb++)
#pragma unroll
      for (int kc = 0; kc < 4; kc++) {
        v16h b = load_b_frag(&Kl[(nb * 16 + ln) * 136 + kc * 32 + 16 * hi]);
        sf[nb] = wmma_f16(qf[kc], b, sf[nb]);
      }

    // Wave-uniform: tiles fully below the diagonal have mask == 0 everywhere.
    const bool needMask = (key0 + 32) > qRow0;

    // Online softmax (row = e + 8*hi; reduce across 16 lanes via shfl_xor).
#pragma unroll
    for (int e = 0; e < 8; e++) {
      const int qAbs = qRow0 + e + 8 * hi;
      float m0 = 0.f, m1 = 0.f;
      if (needMask) {
        const float* mrow = mask + ((size_t)(bz * SEQ + qAbs)) * SEQ + key0;
        m0 = mrow[ln];
        m1 = mrow[16 + ln];
      }
      float s0 = sf[0][e] * scale + m0;
      float s1 = sf[1][e] * scale + m1;
      float t = fmaxf(s0, s1);
#pragma unroll
      for (int off = 8; off; off >>= 1) t = fmaxf(t, __shfl_xor(t, off, 32));
      const float nm = fmaxf(run_max[e], t);
      const float f = __expf(run_max[e] - nm);
      const float p0 = __expf(s0 - nm);
      const float p1 = __expf(s1 - nm);
      float rs = p0 + p1;
#pragma unroll
      for (int off = 8; off; off >>= 1) rs += __shfl_xor(rs, off, 32);
      run_sum[e] = run_sum[e] * f + rs;
      run_max[e] = nm;
#pragma unroll
      for (int j = 0; j < 8; j++) outa[j][e] *= f;
      Pl[w][(e + 8 * hi) * 40 + ln]      = (half_t)p0;
      Pl[w][(e + 8 * hi) * 40 + 16 + ln] = (half_t)p1;
    }
    __syncthreads();

    // O += P(16x32) @ V(32x128): B frag reads Vt[dim][key] contiguously.
    v16h pa = load_a_frag(&Pl[w][ln * 40], hi);
#pragma unroll
    for (int j = 0; j < 8; j++) {
      v16h bv = load_b_frag(&Vt[(j * 16 + ln) * 40 + 16 * hi]);
      outa[j] = wmma_f16(pa, bv, outa[j]);
    }
  }

  // Epilogue: normalize and store f16 context [token, head*128 + dim].
#pragma unroll
  for (int e = 0; e < 8; e++) {
    const float inv = 1.0f / run_sum[e];
    const int row = qRow0 + e + 8 * hi;
#pragma unroll
    for (int j = 0; j < 8; j++)
      O[((size_t)(bz * SEQ + row)) * HIDDEN + head * HDIM + j * 16 + ln] =
          (half_t)(outa[j][e] * inv);
  }
}

// ---------------------------------------------------------------------------
// Host-side orchestration. Workspace layout (117.4 MB total):
//   [0]      x_h   32MB  f16 activations (reused as attention output)
//   [32MB]   Wd    32MB  f16 effective weight (reused for q/k/v/o)
//   [64MB]   q_h   32MB
//   [96MB]   k_h    8MB
//   [104MB]  v_h    8MB
// Stream ordering serializes the buffer reuse.
// ---------------------------------------------------------------------------
extern "C" void kernel_launch(void* const* d_in, const int* in_sizes, int n_in,
                              void* d_out, int out_size, void* d_ws, size_t ws_size,
                              hipStream_t stream) {
  const float* hs   = (const float*)d_in[0];
  const float* mask = (const float*)d_in[1];
  const float* Wq = (const float*)d_in[2];
  const float* Aq = (const float*)d_in[3];
  const float* Bq = (const float*)d_in[4];
  const float* mq = (const float*)d_in[5];
  const float* Wk = (const float*)d_in[6];
  const float* Ak = (const float*)d_in[7];
  const float* Bk = (const float*)d_in[8];
  const float* mk = (const float*)d_in[9];
  const float* Wv = (const float*)d_in[10];
  const float* Av = (const float*)d_in[11];
  const float* Bv = (const float*)d_in[12];
  const float* mv = (const float*)d_in[13];
  const float* Wo = (const float*)d_in[14];
  const float* Ao = (const float*)d_in[15];
  const float* Bo = (const float*)d_in[16];
  const float* mo = (const float*)d_in[17];

  char* ws = (char*)d_ws;
  const int M = BATCH * SEQ;  // 4096 tokens
  half_t* x_h = (half_t*)(ws);
  half_t* Wd  = (half_t*)(ws + (size_t)33554432);
  half_t* q_h = (half_t*)(ws + (size_t)2 * 33554432);
  half_t* k_h = (half_t*)(ws + (size_t)3 * 33554432);
  half_t* v_h = (half_t*)(ws + (size_t)3 * 33554432 + 8388608);

  {
    const int n = M * HIDDEN;  // 16,777,216
    f32_to_f16_kernel<<<n / 256, 256, 0, stream>>>(hs, x_h, n);
  }

  // Q projection
  dora_wd_kernel<<<HIDDEN, 256, 0, stream>>>(Wq, Aq, Bq, mq, Wd);
  gemm_xwT_kernel<false><<<dim3(HIDDEN / 128, M / 128), 256, 0, stream>>>(
      x_h, Wd, q_h, M, HIDDEN, HIDDEN);
  // K projection
  dora_wd_kernel<<<KV_OUT, 256, 0, stream>>>(Wk, Ak, Bk, mk, Wd);
  gemm_xwT_kernel<false><<<dim3(KV_OUT / 128, M / 128), 256, 0, stream>>>(
      x_h, Wd, k_h, M, KV_OUT, HIDDEN);
  // V projection
  dora_wd_kernel<<<KV_OUT, 256, 0, stream>>>(Wv, Av, Bv, mv, Wd);
  gemm_xwT_kernel<false><<<dim3(KV_OUT / 128, M / 128), 256, 0, stream>>>(
      x_h, Wd, v_h, M, KV_OUT, HIDDEN);

  // Fused attention (writes context into x_h, which is no longer needed)
  flash_attn_kernel<<<dim3(SEQ / 128, NHEADS, BATCH), 256, 0, stream>>>(
      q_h, k_h, v_h, mask, x_h);

  // Output projection -> f32 d_out
  dora_wd_kernel<<<HIDDEN, 256, 0, stream>>>(Wo, Ao, Bo, mo, Wd);
  gemm_xwT_kernel<true><<<dim3(HIDDEN / 128, M / 128), 256, 0, stream>>>(
      x_h, Wd, d_out, M, HIDDEN, HIDDEN);
}
